// AnaC2f_pro_31928786878549
// MI455X (gfx1250) — matrix operand order repaired and checked by reading, not compiled, not run on hardware
//
#include <hip/hip_runtime.h>
#include <hip/hip_bf16.h>
#include <stdint.h>

typedef __attribute__((ext_vector_type(16))) _Float16 v16h;
typedef __attribute__((ext_vector_type(8)))  _Float16 v8h;
typedef __attribute__((ext_vector_type(8)))  float    v8f;
typedef __attribute__((ext_vector_type(2)))  float    v2f;

static constexpr int kB = 16, kC = 128, kHW = 25600, kS = 256, kN = 4096;

#if defined(__has_builtin)
#if __has_builtin(__builtin_amdgcn_wmma_f32_16x16x4_f32)
#define HAVE_WMMA_F32X4 1
#endif
#endif

__device__ __forceinline__ unsigned fkey(float f) {
  unsigned u = __float_as_uint(f);
  return (u & 0x80000000u) ? ~u : (u | 0x80000000u);
}

// ---------------- passthrough copy of z -> out (float4, 420 MB total traffic) ----
__global__ void copy_kernel(const float4* __restrict__ src, float4* __restrict__ dst, int n4) {
  int i = blockIdx.x * blockDim.x + threadIdx.x;
  if (i < n4) dst[i] = src[i];
}

__global__ void zero_kernel(float* __restrict__ p, int n) {
  int i = blockIdx.x * blockDim.x + threadIdx.x;
  if (i < n) p[i] = 0.0f;
}

// ---------------- deterministic per-batch top-256 (4-pass radix select) --------
__global__ __launch_bounds__(1024) void topk_kernel(const float* __restrict__ score,
                                                    int* __restrict__ top_idx) {
  const int b = blockIdx.x;
  const float* s = score + (size_t)b * kHW;
  __shared__ unsigned hist[256];
  __shared__ unsigned sel_prefix, sel_remaining;
  __shared__ unsigned cnt_gt, cnt_eq;
  const int t = threadIdx.x;

  unsigned prefix = 0;     // value of (key >> shift_next_group) selected so far
  unsigned remaining = kS; // how many still needed within current prefix class
  for (int pass = 0; pass < 4; ++pass) {
    const int shift = 24 - pass * 8;
    if (t < 256) hist[t] = 0;
    __syncthreads();
    for (int e = t; e < kHW; e += 1024) {
      unsigned k = fkey(s[e]);
      unsigned kk = k >> shift;
      if ((kk >> 8) == prefix) atomicAdd(&hist[kk & 0xFFu], 1u);
    }
    __syncthreads();
    if (t == 0) {
      unsigned cum = 0; int d = 255;
      for (; d > 0; --d) { cum += hist[d]; if (cum >= remaining) break; }
      if (cum < remaining) cum += hist[0]; // d==0 fallthrough
      sel_prefix = (prefix << 8) | (unsigned)d;
      sel_remaining = remaining - (cum - hist[d]);
    }
    __syncthreads();
    prefix = sel_prefix;
    remaining = sel_remaining;
    __syncthreads();
  }
  const unsigned T = prefix;          // exact 32-bit threshold key
  const unsigned need_eq = remaining; // ties still needed at == T
  if (t == 0) { cnt_gt = 0; cnt_eq = 0; }
  __syncthreads();
  for (int e = t; e < kHW; e += 1024) {
    unsigned k = fkey(s[e]);
    if (k > T) {
      unsigned p = atomicAdd(&cnt_gt, 1u);
      top_idx[b * kS + p] = e;
    } else if (k == T) {
      unsigned p = atomicAdd(&cnt_eq, 1u);
      if (p < need_eq) top_idx[b * kS + (kS - need_eq) + p] = e;
    }
  }
}

// ---------------- gather selected features + row-normalize to f16 -------------
__global__ __launch_bounds__(128) void gather_kernel(const float* __restrict__ z,
                                                     const int* __restrict__ top_idx,
                                                     float* __restrict__ feats,
                                                     _Float16* __restrict__ nfh) {
  __shared__ float red[128];
  const int row = blockIdx.x;       // 0..N-1
  const int b = row / kS;
  const int c = threadIdx.x;
  const int idx = top_idx[row];
  const float v = z[((size_t)b * kC + c) * kHW + idx];
  feats[row * kC + c] = v;
  red[c] = v * v;
  __syncthreads();
  for (int s2 = 64; s2 > 0; s2 >>= 1) {
    if (c < s2) red[c] += red[c + s2];
    __syncthreads();
  }
  const float rn = 1.0f / fmaxf(sqrtf(red[0]), 1e-12f);
  nfh[row * kC + c] = (_Float16)(v * rn);
}

// ---------------- sim = nf @ nf^T (WMMA f16), threshold -> adjacency + degree --
// one wave per 16x16 tile; A frag = one contiguous 32B load, B frag = one contiguous 32B load
__global__ __launch_bounds__(256) void sim_kernel(const _Float16* __restrict__ nfh,
                                                  _Float16* __restrict__ adjh,
                                                  float* __restrict__ deg) {
  const int lane = threadIdx.x & 31;
  const int gw = blockIdx.x * 8 + (threadIdx.x >> 5);
  const int it = gw >> 8;          // 0..255
  const int jt = gw & 255;         // 0..255
  const int m = lane & 15;
  const int half = lane >> 4;

  v8f acc = {};
  const _Float16* arow = nfh + (size_t)(it * 16 + m) * kC;
  const _Float16* brow = nfh + (size_t)(jt * 16 + m) * kC; // B column n = m
#pragma unroll
  for (int kc = 0; kc < kC; kc += 32) {
    v16h a = *(const v16h*)(arow + kc + half * 16);
    v16h bfr = *(const v16h*)(brow + kc + half * 16);
    acc = __builtin_amdgcn_wmma_f32_16x16x32_f16(false, a, false, bfr, (short)0, acc,
                                                 false, false);
  }
#pragma unroll
  for (int r = 0; r < 8; ++r) {
    const int i = it * 16 + half * 8 + r;
    const int j = jt * 16 + m;
    float adjv = (acc[r] > 0.6f) ? 1.0f : 0.0f; // cos sim > 0.6 (includes self loop)
    adjh[(size_t)i * kN + j] = (_Float16)adjv;
    float rs = adjv;
    rs += __shfl_xor(rs, 1, 16);
    rs += __shfl_xor(rs, 2, 16);
    rs += __shfl_xor(rs, 4, 16);
    rs += __shfl_xor(rs, 8, 16);
    if (m == 0) atomicAdd(&deg[i], rs); // sums of 0/1 -> exact & order-independent
  }
}

// ---------------- dinv + transposed scaled features XdT[c][i] ------------------
__global__ void xd_kernel(const float* __restrict__ deg, const float* __restrict__ feats,
                          float* __restrict__ dinv, _Float16* __restrict__ xdT) {
  int t = blockIdx.x * blockDim.x + threadIdx.x;
  if (t >= kN * kC) return;
  const int i = t / kC, c = t % kC;
  const float dv = rsqrtf(fmaxf(deg[i], 1.0f));
  if (c == 0) dinv[i] = dv;
  xdT[(size_t)c * kN + i] = (_Float16)(dv * feats[t]);
}

// ---------------- agg = dinv * (A @ (dinv*X))  (WMMA f16, K = 4096) ------------
__global__ __launch_bounds__(256) void agg_kernel(const _Float16* __restrict__ adjh,
                                                  const _Float16* __restrict__ xdT,
                                                  const float* __restrict__ dinv,
                                                  float* __restrict__ aggF) {
  const int lane = threadIdx.x & 31;
  const int gw = blockIdx.x * 8 + (threadIdx.x >> 5);
  const int it = gw >> 3;          // 0..255
  const int ct = gw & 7;           // 0..7
  const int m = lane & 15;
  const int half = lane >> 4;

  v8f acc = {};
  const _Float16* arow = adjh + (size_t)(it * 16 + m) * kN;
  const _Float16* brow = xdT + (size_t)(ct * 16 + m) * kN;
  for (int kc = 0; kc < kN; kc += 32) {
    v16h a = *(const v16h*)(arow + kc + half * 16);
    v16h bfr = *(const v16h*)(brow + kc + half * 16);
    acc = __builtin_amdgcn_wmma_f32_16x16x32_f16(false, a, false, bfr, (short)0, acc,
                                                 false, false);
  }
#pragma unroll
  for (int r = 0; r < 8; ++r) {
    const int i = it * 16 + half * 8 + r;
    aggF[(size_t)i * kC + ct * 16 + m] = dinv[i] * acc[r];
  }
}

// ---------------- upd = agg @ W + b  (f32 WMMA 16x16x4, exact f32) -------------
__global__ __launch_bounds__(256) void proj_kernel(const float* __restrict__ aggF,
                                                   const float* __restrict__ Wg,
                                                   const float* __restrict__ bg,
                                                   float* __restrict__ upd) {
  const int lane = threadIdx.x & 31;
  const int gw = blockIdx.x * 8 + (threadIdx.x >> 5);
  const int it = gw >> 3;
  const int ct = gw & 7;
  const int m = lane & 15;
  const int half = lane >> 4;
  const int col = ct * 16 + m;
#if HAVE_WMMA_F32X4
  v8f acc = {};
  const float* arow = aggF + (size_t)(it * 16 + m) * kC;
#pragma unroll
  for (int kc = 0; kc < kC; kc += 4) {
    v2f a = *(const v2f*)(arow + kc + half * 2);
    v2f bfr;
    bfr[0] = Wg[(kc + half * 2 + 0) * kC + col];
    bfr[1] = Wg[(kc + half * 2 + 1) * kC + col];
    acc = __builtin_amdgcn_wmma_f32_16x16x4_f32(false, a, false, bfr, (short)0, acc,
                                                false, false);
  }
#pragma unroll
  for (int r = 0; r < 8; ++r) {
    const int i = it * 16 + half * 8 + r;
    upd[(size_t)i * kC + col] = acc[r] + bg[col];
  }
#else
  // scalar fallback (0.13 GFLOP total; negligible)
  for (int r = 0; r < 8; ++r) {
    const int i = it * 16 + half * 8 + r;
    float s = bg[col];
    for (int k = 0; k < kC; ++k) s += aggF[(size_t)i * kC + k] * Wg[k * kC + col];
    upd[(size_t)i * kC + col] = s;
  }
#endif
}

// ---------------- scatter updated rows back into out ---------------------------
__global__ void scatter_kernel(const float* __restrict__ upd,
                               const int* __restrict__ top_idx,
                               float* __restrict__ out) {
  int t = blockIdx.x * blockDim.x + threadIdx.x;
  if (t >= kN * kC) return;
  const int row = t / kC, c = t % kC;
  const int b = row / kS;
  const int idx = top_idx[row];
  out[((size_t)b * kC + c) * kHW + idx] = upd[t];
}

extern "C" void kernel_launch(void* const* d_in, const int* in_sizes, int n_in,
                              void* d_out, int out_size, void* d_ws, size_t ws_size,
                              hipStream_t stream) {
  (void)in_sizes; (void)n_in; (void)out_size; (void)ws_size;
  const float* z     = (const float*)d_in[0];   // [16,128,160,160]
  const float* score = (const float*)d_in[1];   // [16,160,160]
  const float* Wg    = (const float*)d_in[2];   // [128,128]
  const float* bg    = (const float*)d_in[3];   // [128]
  float* out = (float*)d_out;

  uintptr_t w = (uintptr_t)d_ws;
  auto carve = [&](size_t bytes) -> void* {
    uintptr_t p = (w + 255) & ~(uintptr_t)255;
    w = p + bytes;
    return (void*)p;
  };
  int*      top_idx = (int*)carve((size_t)kN * sizeof(int));       // 16 KB
  float*    feats   = (float*)carve((size_t)kN * kC * 4);          // 2 MB
  _Float16* nfh     = (_Float16*)carve((size_t)kN * kC * 2);       // 1 MB
  _Float16* adjh    = (_Float16*)carve((size_t)kN * kN * 2);       // 32 MB (L2-resident)
  float*    deg     = (float*)carve((size_t)kN * 4);
  float*    dinv    = (float*)carve((size_t)kN * 4);
  _Float16* xdT     = (_Float16*)carve((size_t)kN * kC * 2);       // 1 MB
  float*    aggF    = (float*)carve((size_t)kN * kC * 4);          // 2 MB
  float*    upd     = (float*)carve((size_t)kN * kC * 4);          // 2 MB

  const int n4 = kB * kC * kHW / 4;
  copy_kernel<<<(n4 + 255) / 256, 256, 0, stream>>>((const float4*)z, (float4*)out, n4);
  topk_kernel<<<kB, 1024, 0, stream>>>(score, top_idx);
  gather_kernel<<<kN, 128, 0, stream>>>(z, top_idx, feats, nfh);
  zero_kernel<<<(kN + 255) / 256, 256, 0, stream>>>(deg, kN);
  sim_kernel<<<(256 * 256) / 8, 256, 0, stream>>>(nfh, adjh, deg);
  xd_kernel<<<(kN * kC + 255) / 256, 256, 0, stream>>>(deg, feats, dinv, xdT);
  agg_kernel<<<(256 * 8) / 8, 256, 0, stream>>>(adjh, xdT, dinv, aggF);
  proj_kernel<<<(256 * 8) / 8, 256, 0, stream>>>(aggF, Wg, bg, upd);
  scatter_kernel<<<(kN * kC + 255) / 256, 256, 0, stream>>>(upd, top_idx, out);
}
